// myYOLO_26594437497623
// MI455X (gfx1250) — compile-verified, matrix operand order checked by reading, not run on hardware
//
#include <hip/hip_runtime.h>
#include <stdint.h>

// ---------------- problem constants ----------------
#define HS 40
#define WSZ 40
#define HW 1600
#define NCLS 20
#define CONF_TH 0.01f
#define NMS_TH 0.5f

// ---------------- types for WMMA ----------------
typedef __attribute__((ext_vector_type(16))) __bf16 v16bf;
typedef __attribute__((ext_vector_type(8)))  __bf16 v8bf;
typedef __attribute__((ext_vector_type(8)))  float  v8f;
typedef __attribute__((ext_vector_type(8)))  unsigned short v8us;

__device__ __forceinline__ unsigned short f2bf(float f) {
    unsigned int u = __float_as_uint(f);
    u += 0x7FFFu + ((u >> 16) & 1u);           // round-to-nearest-even
    return (unsigned short)(u >> 16);
}

// Per-lane 16-bit WMMA operand tile: two 16B chunks at +0 and +32 bytes.
__device__ __forceinline__ v16bf ldtile(const unsigned short* p) {
    v8bf lo = *(const v8bf*)(p);
    v8bf hi = *(const v8bf*)(p + 16);
    return __builtin_shufflevector(lo, hi, 0,1,2,3,4,5,6,7,8,9,10,11,12,13,14,15);
}

__device__ __forceinline__ v8f wmma_bf16(v16bf a, v16bf b, v8f c) {
    return __builtin_amdgcn_wmma_f32_16x16x32_bf16(false, a, false, b, (short)0, c, false, false);
}

// D-tile store: lane holds column n, rows m0..m0+7 contiguous.
__device__ __forceinline__ void store_bf(v8f acc, const float* __restrict__ bias,
                                         unsigned short* __restrict__ out, int n, int m0, int M) {
    v8us pk;
#pragma unroll
    for (int i = 0; i < 8; i++) {
        float v = acc[i] + bias[m0 + i];
        v = v > 0.f ? v : 0.1f * v;                 // LeakyReLU(0.1)
        pk[i] = f2bf(v);
    }
    *(v8us*)(out + (size_t)n * M + m0) = pk;
}

__device__ __forceinline__ void store_f32(v8f acc, const float* __restrict__ bias,
                                          float* __restrict__ out, int n, int m0, int M) {
    float* orow = out + (size_t)n * M + m0;
#pragma unroll
    for (int i = 0; i < 8; i++) orow[i] = acc[i] + bias[m0 + i];
}

// ---------------- SPP: NCHW f32 -> [pixel][2048] bf16 ----------------
__global__ void spp_kernel(const float* __restrict__ feat, unsigned short* __restrict__ X0) {
    int idx = blockIdx.x * blockDim.x + threadIdx.x;   // p*512 + c
    if (idx >= HW * 512) return;
    int p = idx >> 9;
    int c = idx & 511;
    int y = p / WSZ, x = p - y * WSZ;
    const float* fc = feat + (size_t)c * HW;
    float ctr = fc[p];
    float m5 = -__builtin_inff(), m9 = -__builtin_inff(), m13 = -__builtin_inff();
    for (int dy = -6; dy <= 6; dy++) {
        int py = y + dy; if (py < 0 || py >= HS) continue;
        for (int dx = -6; dx <= 6; dx++) {
            int px = x + dx; if (px < 0 || px >= WSZ) continue;
            float v = fc[py * WSZ + px];
            m13 = fmaxf(m13, v);
            if (dy >= -4 && dy <= 4 && dx >= -4 && dx <= 4) m9 = fmaxf(m9, v);
            if (dy >= -2 && dy <= 2 && dx >= -2 && dx <= 2) m5 = fmaxf(m5, v);
        }
    }
    unsigned short* row = X0 + (size_t)p * 2048;
    row[c]          = f2bf(ctr);
    row[512 + c]    = f2bf(m5);
    row[1024 + c]   = f2bf(m9);
    row[1536 + c]   = f2bf(m13);
}

// ------- weight repack: OIHW f32 -> [co][tap*Cin+ci] bf16 with BN folded -------
__global__ void repack_kernel(const float* __restrict__ w, const float* __restrict__ gamma,
                              const float* __restrict__ beta, const float* __restrict__ mean,
                              const float* __restrict__ var,
                              unsigned short* __restrict__ wOut, float* __restrict__ bOut,
                              int Cout, int Cin, int K2) {
    int idx = blockIdx.x * blockDim.x + threadIdx.x;
    int total = Cout * Cin * K2;
    if (idx >= total) return;
    int co = idx / (Cin * K2);
    int r  = idx - co * (Cin * K2);
    int ci = r / K2;
    int k  = r - ci * K2;
    float scale = gamma[co] * rsqrtf(var[co] + 1e-5f);
    wOut[(size_t)co * (Cin * K2) + (size_t)k * Cin + ci] = f2bf(w[idx] * scale);
    if (r == 0) bOut[co] = beta[co] - mean[co] * scale;
}

__global__ void pred_repack(const float* __restrict__ w, const float* __restrict__ b,
                            unsigned short* __restrict__ wOut, float* __restrict__ bOut) {
    int idx = blockIdx.x * blockDim.x + threadIdx.x;   // 32*512
    if (idx >= 32 * 512) return;
    int co = idx >> 9, ci = idx & 511;
    float v = (co < 25) ? w[co * 512 + ci] : 0.f;
    wOut[idx] = f2bf(v);
    if (ci == 0) bOut[co] = (co < 25) ? b[co] : 0.f;
}

// zero pad rows (row index HW) of the 3x3 input activations
__global__ void zero_pad_rows(unsigned short* __restrict__ X2, unsigned short* __restrict__ X4) {
    int i = blockIdx.x * blockDim.x + threadIdx.x;
    if (i < 256) {
        X2[(size_t)HW * 256 + i] = 0;
        X4[(size_t)HW * 256 + i] = 0;
    }
}

// ---------------- 1x1 conv as WMMA GEMM ----------------
// out[p][m] = act( sum_k W[m][k]*X[p][k] + bias[m] )
// MODE 0: bf16 out + LeakyReLU.  MODE 1: f32 out, no activation.
// K must be a multiple of 64.
template <int MODE>
__global__ void gemm1x1(const unsigned short* __restrict__ W,
                        const unsigned short* __restrict__ X,
                        const float* __restrict__ bias,
                        unsigned short* __restrict__ outBf,
                        float* __restrict__ outF,
                        int M, int K) {
    const int lane = threadIdx.x & 31;
    const int wave = blockIdx.x * (blockDim.x >> 5) + (threadIdx.x >> 5);
    const int mt   = wave / 25;
    const int ntb  = wave - mt * 25;
    if (mt >= (M >> 4)) return;
    const int l15   = lane & 15;
    const int khalf = lane >> 4;
    const int m     = mt * 16 + l15;

    v8f a0 = {}, a1 = {}, a2 = {}, a3 = {};
    const unsigned short* wp  = W + (size_t)m * K + khalf * 8;
    const unsigned short* xp0 = X + (size_t)(ntb * 64 +  0 + l15) * K + khalf * 8;
    const unsigned short* xp1 = X + (size_t)(ntb * 64 + 16 + l15) * K + khalf * 8;
    const unsigned short* xp2 = X + (size_t)(ntb * 64 + 32 + l15) * K + khalf * 8;
    const unsigned short* xp3 = X + (size_t)(ntb * 64 + 48 + l15) * K + khalf * 8;

#pragma unroll 1
    for (int k0 = 0; k0 < K; k0 += 64) {
        // half 1 (k0) and half 2 (k0+32) loads issued together, WMMAs follow:
        v16bf avA = ldtile(wp);
        v16bf b0A = ldtile(xp0);
        v16bf b1A = ldtile(xp1);
        v16bf b2A = ldtile(xp2);
        v16bf b3A = ldtile(xp3);
        v16bf avB = ldtile(wp + 32);
        v16bf b0B = ldtile(xp0 + 32);
        v16bf b1B = ldtile(xp1 + 32);
        v16bf b2B = ldtile(xp2 + 32);
        v16bf b3B = ldtile(xp3 + 32);
        a0 = wmma_bf16(avA, b0A, a0);
        a1 = wmma_bf16(avA, b1A, a1);
        a2 = wmma_bf16(avA, b2A, a2);
        a3 = wmma_bf16(avA, b3A, a3);
        a0 = wmma_bf16(avB, b0B, a0);
        a1 = wmma_bf16(avB, b1B, a1);
        a2 = wmma_bf16(avB, b2B, a2);
        a3 = wmma_bf16(avB, b3B, a3);
        wp += 64; xp0 += 64; xp1 += 64; xp2 += 64; xp3 += 64;
    }

    const int m0 = mt * 16 + khalf * 8;
    const int nb = ntb * 64 + l15;
    if (MODE == 0) {
        store_bf(a0, bias, outBf, nb +  0, m0, M);
        store_bf(a1, bias, outBf, nb + 16, m0, M);
        store_bf(a2, bias, outBf, nb + 32, m0, M);
        store_bf(a3, bias, outBf, nb + 48, m0, M);
    } else {
        store_f32(a0, bias, outF, nb +  0, m0, M);
        store_f32(a1, bias, outF, nb + 16, m0, M);
        store_f32(a2, bias, outF, nb + 32, m0, M);
        store_f32(a3, bias, outF, nb + 48, m0, M);
    }
}

// ---------------- 3x3 conv (pad 1) as WMMA GEMM, M=512, Cin=256, K=2304 ----------------
// X: [(HW+1)][256] bf16, row HW is all zeros (border fallback).
__global__ void gemm3x3(const unsigned short* __restrict__ W,
                        const unsigned short* __restrict__ X,
                        const float* __restrict__ bias,
                        unsigned short* __restrict__ outBf) {
    const int lane = threadIdx.x & 31;
    const int wave = blockIdx.x * (blockDim.x >> 5) + (threadIdx.x >> 5);
    const int mt   = wave / 25;                 // 0..31
    const int ntb  = wave - mt * 25;
    const int l15   = lane & 15;
    const int khalf = lane >> 4;
    const int m     = mt * 16 + l15;

    int yy0, xx0, yy1, xx1, yy2, xx2, yy3, xx3;
    {
        int p0 = ntb * 64 +  0 + l15; yy0 = p0 / WSZ; xx0 = p0 - yy0 * WSZ;
        int p1 = ntb * 64 + 16 + l15; yy1 = p1 / WSZ; xx1 = p1 - yy1 * WSZ;
        int p2 = ntb * 64 + 32 + l15; yy2 = p2 / WSZ; xx2 = p2 - yy2 * WSZ;
        int p3 = ntb * 64 + 48 + l15; yy3 = p3 / WSZ; xx3 = p3 - yy3 * WSZ;
    }

    v8f a0 = {}, a1 = {}, a2 = {}, a3 = {};
    const unsigned short* wp = W + (size_t)m * 2304 + khalf * 8;

#pragma unroll 1
    for (int dy = -1; dy <= 1; dy++) {
#pragma unroll 1
        for (int dx = -1; dx <= 1; dx++) {
            int py0 = yy0 + dy, px0 = xx0 + dx;
            int py1 = yy1 + dy, px1 = xx1 + dx;
            int py2 = yy2 + dy, px2 = xx2 + dx;
            int py3 = yy3 + dy, px3 = xx3 + dx;
            int r0 = ((unsigned)py0 < (unsigned)HS && (unsigned)px0 < (unsigned)WSZ) ? (py0 * WSZ + px0) : HW;
            int r1 = ((unsigned)py1 < (unsigned)HS && (unsigned)px1 < (unsigned)WSZ) ? (py1 * WSZ + px1) : HW;
            int r2 = ((unsigned)py2 < (unsigned)HS && (unsigned)px2 < (unsigned)WSZ) ? (py2 * WSZ + px2) : HW;
            int r3 = ((unsigned)py3 < (unsigned)HS && (unsigned)px3 < (unsigned)WSZ) ? (py3 * WSZ + px3) : HW;
            const unsigned short* xp0 = X + (size_t)r0 * 256 + khalf * 8;
            const unsigned short* xp1 = X + (size_t)r1 * 256 + khalf * 8;
            const unsigned short* xp2 = X + (size_t)r2 * 256 + khalf * 8;
            const unsigned short* xp3 = X + (size_t)r3 * 256 + khalf * 8;

#pragma unroll 1
            for (int c0 = 0; c0 < 256; c0 += 64) {
                v16bf avA = ldtile(wp);
                v16bf b0A = ldtile(xp0);
                v16bf b1A = ldtile(xp1);
                v16bf b2A = ldtile(xp2);
                v16bf b3A = ldtile(xp3);
                v16bf avB = ldtile(wp + 32);
                v16bf b0B = ldtile(xp0 + 32);
                v16bf b1B = ldtile(xp1 + 32);
                v16bf b2B = ldtile(xp2 + 32);
                v16bf b3B = ldtile(xp3 + 32);
                a0 = wmma_bf16(avA, b0A, a0);
                a1 = wmma_bf16(avA, b1A, a1);
                a2 = wmma_bf16(avA, b2A, a2);
                a3 = wmma_bf16(avA, b3A, a3);
                a0 = wmma_bf16(avB, b0B, a0);
                a1 = wmma_bf16(avB, b1B, a1);
                a2 = wmma_bf16(avB, b2B, a2);
                a3 = wmma_bf16(avB, b3B, a3);
                wp += 64; xp0 += 64; xp1 += 64; xp2 += 64; xp3 += 64;
            }
        }
    }

    const int m0 = mt * 16 + khalf * 8;
    const int nb = ntb * 64 + l15;
    store_bf(a0, bias, outBf, nb +  0, m0, 512);
    store_bf(a1, bias, outBf, nb + 16, m0, 512);
    store_bf(a2, bias, outBf, nb + 32, m0, 512);
    store_bf(a3, bias, outBf, nb + 48, m0, 512);
}

// ---------------- decode + score ----------------
__global__ void decode_kernel(const float* __restrict__ pred, float* __restrict__ out,
                              float* __restrict__ boxes, float* __restrict__ scores,
                              int* __restrict__ labels) {
    int p = blockIdx.x * blockDim.x + threadIdx.x;
    if (p >= HW) return;
    const float* P = pred + (size_t)p * 32;
    float conf = P[0];
    float cmax = -__builtin_inff(); int lab = 0;
    for (int i = 0; i < NCLS; i++) { float v = P[1 + i]; if (v > cmax) { cmax = v; lab = i; } }
    float ssum = 0.f;
    for (int i = 0; i < NCLS; i++) ssum += expf(P[1 + i] - cmax);
    float sig  = 1.f / (1.f + expf(-conf));
    float msc  = (1.f / ssum) * sig;

    float tx = P[21], ty = P[22], tw = P[23], th = P[24];
    int gy = p / WSZ, gx = p - gy * WSZ;
    float cx = (1.f / (1.f + expf(-tx)) + (float)gx) * 32.f;
    float cy = (1.f / (1.f + expf(-ty)) + (float)gy) * 32.f;
    float bw = expf(tw), bh = expf(th);
    const float inv = 1.f / 1280.f;
    float x1 = fminf(fmaxf((cx - 0.5f * bw) * inv, 0.f), 1.f);
    float y1 = fminf(fmaxf((cy - 0.5f * bh) * inv, 0.f), 1.f);
    float x2 = fminf(fmaxf((cx + 0.5f * bw) * inv, 0.f), 1.f);
    float y2 = fminf(fmaxf((cy + 0.5f * bh) * inv, 0.f), 1.f);

    boxes[p * 4 + 0] = x1; boxes[p * 4 + 1] = y1;
    boxes[p * 4 + 2] = x2; boxes[p * 4 + 3] = y2;
    scores[p] = msc; labels[p] = lab;

    out[p * 4 + 0] = x1; out[p * 4 + 1] = y1;
    out[p * 4 + 2] = x2; out[p * 4 + 3] = y2;
    out[4 * HW + p] = msc;
    out[5 * HW + p] = (float)lab;
}

// ---------------- IoU matrix ----------------
__global__ void iou_kernel(const float* __restrict__ boxes, float* __restrict__ iou) {
    int idx = blockIdx.x * blockDim.x + threadIdx.x;
    if (idx >= HW * HW) return;
    int i = idx / HW, j = idx - i * HW;
    float x1i = boxes[i*4+0], y1i = boxes[i*4+1], x2i = boxes[i*4+2], y2i = boxes[i*4+3];
    float x1j = boxes[j*4+0], y1j = boxes[j*4+1], x2j = boxes[j*4+2], y2j = boxes[j*4+3];
    float ai = (x2i - x1i) * (y2i - y1i);
    float aj = (x2j - x1j) * (y2j - y1j);
    float w = fmaxf(1e-10f, fminf(x2i, x2j) - fmaxf(x1i, x1j));
    float h = fmaxf(1e-10f, fminf(y2i, y2j) - fmaxf(y1i, y1j));
    float inter = w * h;
    iou[idx] = inter / (ai + aj - inter);
}

// ---------------- stable descending rank sort (matches argsort(-scores)) ----------------
__global__ void rank_kernel(const float* __restrict__ scores, int* __restrict__ order) {
    int j = blockIdx.x * blockDim.x + threadIdx.x;
    if (j >= HW) return;
    float sj = scores[j];
    int r = 0;
    for (int i = 0; i < HW; i++) {
        float si = scores[i];
        r += (si > sj) || (si == sj && i < j);
    }
    order[r] = j;
}

__global__ void zero_keep(int* __restrict__ keepInt) {
    int j = blockIdx.x * blockDim.x + threadIdx.x;
    if (j < HW) keepInt[j] = 0;
}

// ---------------- per-class greedy NMS, one block per class ----------------
__global__ void nms_kernel(const int* __restrict__ order, const int* __restrict__ labels,
                           const float* __restrict__ scores, const float* __restrict__ iou,
                           int* __restrict__ keepInt) {
    const int cls = blockIdx.x;
    __shared__ unsigned char sup[HW];
    __shared__ int sh_cur, sh_act;
    for (int j = threadIdx.x; j < HW; j += blockDim.x) sup[j] = 0;
    __syncthreads();
    for (int i = 0; i < HW; i++) {
        if (threadIdx.x == 0) {
            int cur = order[i];
            int act = (labels[cur] == cls) && (scores[cur] >= CONF_TH) && (!sup[cur]);
            sh_cur = cur; sh_act = act;
            if (act) atomicOr(&keepInt[cur], 1);
        }
        __syncthreads();
        if (sh_act) {
            const float* irow = iou + (size_t)sh_cur * HW;
            for (int jp = i + 1 + (int)threadIdx.x; jp < HW; jp += blockDim.x) {
                int j = order[jp];
                if (irow[j] > NMS_TH) sup[j] = 1;
            }
        }
        __syncthreads();
    }
}

__global__ void finalize_keep(const int* __restrict__ keepInt, float* __restrict__ out) {
    int j = blockIdx.x * blockDim.x + threadIdx.x;
    if (j < HW) out[6 * HW + j] = keepInt[j] ? 1.f : 0.f;
}

// ---------------- workspace layout ----------------
static constexpr size_t AL(size_t x) { return (x + 255) & ~(size_t)255; }
static constexpr size_t OFF_X0 = 0;                                  // [1600][2048] bf16
static constexpr size_t OFF_WN = AL(OFF_X0 + (size_t)HW*2048*2);     // [512][2048] bf16
static constexpr size_t OFF_BN = AL(OFF_WN + (size_t)512*2048*2);
static constexpr size_t OFF_X1 = AL(OFF_BN + 512*4);                 // [1600][512] bf16
static constexpr size_t OFF_IOU = 0;                                 // aliases X0..X1 (dead by IoU time)
static constexpr size_t END_X1 = OFF_X1 + (size_t)HW*512*2;
static_assert(END_X1 >= (size_t)HW*HW*4, "IoU alias region too small");
static constexpr size_t OFF_W1 = AL(END_X1);                         // [256][512] bf16
static constexpr size_t OFF_B1 = AL(OFF_W1 + (size_t)256*512*2);
static constexpr size_t OFF_X2 = AL(OFF_B1 + 256*4);                 // [(1600+1)][256] bf16 (+zero row)
static constexpr size_t OFF_W2 = AL(OFF_X2 + (size_t)(HW+1)*256*2);  // [512][2304] bf16
static constexpr size_t OFF_B2 = AL(OFF_W2 + (size_t)512*2304*2);
static constexpr size_t OFF_X3 = AL(OFF_B2 + 512*4);                 // [1600][512] bf16
static constexpr size_t OFF_W3 = AL(OFF_X3 + (size_t)HW*512*2);      // [256][512] bf16
static constexpr size_t OFF_B3 = AL(OFF_W3 + (size_t)256*512*2);
static constexpr size_t OFF_X4 = AL(OFF_B3 + 256*4);                 // [(1600+1)][256] bf16 (+zero row)
static constexpr size_t OFF_W4 = AL(OFF_X4 + (size_t)(HW+1)*256*2);  // [512][2304] bf16
static constexpr size_t OFF_B4 = AL(OFF_W4 + (size_t)512*2304*2);
static constexpr size_t OFF_X5 = AL(OFF_B4 + 512*4);                 // [1600][512] bf16
static constexpr size_t OFF_WP = AL(OFF_X5 + (size_t)HW*512*2);      // [32][512] bf16
static constexpr size_t OFF_BP = AL(OFF_WP + (size_t)32*512*2);
static constexpr size_t OFF_PR = AL(OFF_BP + 32*4);                  // [1600][32] f32
static constexpr size_t OFF_BX = AL(OFF_PR + (size_t)HW*32*4);       // [1600][4] f32
static constexpr size_t OFF_SC = AL(OFF_BX + (size_t)HW*4*4);
static constexpr size_t OFF_LB = AL(OFF_SC + (size_t)HW*4);
static constexpr size_t OFF_OR = AL(OFF_LB + (size_t)HW*4);
static constexpr size_t OFF_KP = AL(OFF_OR + (size_t)HW*4);
static constexpr size_t WS_NEED = AL(OFF_KP + (size_t)HW*4);

extern "C" void kernel_launch(void* const* d_in, const int* in_sizes, int n_in,
                              void* d_out, int out_size, void* d_ws, size_t ws_size,
                              hipStream_t stream) {
    (void)in_sizes; (void)n_in; (void)out_size; (void)ws_size; (void)WS_NEED;
    const float* feat   = (const float*)d_in[0];
    const float* nW     = (const float*)d_in[1];
    const float* nG     = (const float*)d_in[2];
    const float* nB     = (const float*)d_in[3];
    const float* nM     = (const float*)d_in[4];
    const float* nV     = (const float*)d_in[5];
    const float* c1W    = (const float*)d_in[6];
    const float* c1G    = (const float*)d_in[7];
    const float* c1B    = (const float*)d_in[8];
    const float* c1M    = (const float*)d_in[9];
    const float* c1V    = (const float*)d_in[10];
    const float* c2W    = (const float*)d_in[11];
    const float* c2G    = (const float*)d_in[12];
    const float* c2B    = (const float*)d_in[13];
    const float* c2M    = (const float*)d_in[14];
    const float* c2V    = (const float*)d_in[15];
    const float* c3W    = (const float*)d_in[16];
    const float* c3G    = (const float*)d_in[17];
    const float* c3B    = (const float*)d_in[18];
    const float* c3M    = (const float*)d_in[19];
    const float* c3V    = (const float*)d_in[20];
    const float* c4W    = (const float*)d_in[21];
    const float* c4G    = (const float*)d_in[22];
    const float* c4B    = (const float*)d_in[23];
    const float* c4M    = (const float*)d_in[24];
    const float* c4V    = (const float*)d_in[25];
    const float* pW     = (const float*)d_in[26];
    const float* pB     = (const float*)d_in[27];

    char* ws = (char*)d_ws;
    unsigned short* X0 = (unsigned short*)(ws + OFF_X0);
    unsigned short* WN = (unsigned short*)(ws + OFF_WN);
    float*          BN = (float*)(ws + OFF_BN);
    unsigned short* X1 = (unsigned short*)(ws + OFF_X1);
    unsigned short* W1 = (unsigned short*)(ws + OFF_W1);
    float*          B1 = (float*)(ws + OFF_B1);
    unsigned short* X2 = (unsigned short*)(ws + OFF_X2);
    unsigned short* W2 = (unsigned short*)(ws + OFF_W2);
    float*          B2 = (float*)(ws + OFF_B2);
    unsigned short* X3 = (unsigned short*)(ws + OFF_X3);
    unsigned short* W3 = (unsigned short*)(ws + OFF_W3);
    float*          B3 = (float*)(ws + OFF_B3);
    unsigned short* X4 = (unsigned short*)(ws + OFF_X4);
    unsigned short* W4 = (unsigned short*)(ws + OFF_W4);
    float*          B4 = (float*)(ws + OFF_B4);
    unsigned short* X5 = (unsigned short*)(ws + OFF_X5);
    unsigned short* WP = (unsigned short*)(ws + OFF_WP);
    float*          BP = (float*)(ws + OFF_BP);
    float*          PR = (float*)(ws + OFF_PR);
    float*          BX = (float*)(ws + OFF_BX);
    float*          SC = (float*)(ws + OFF_SC);
    int*            LB = (int*)(ws + OFF_LB);
    int*            OR_ = (int*)(ws + OFF_OR);
    int*            KP = (int*)(ws + OFF_KP);
    float*          IOU = (float*)(ws + OFF_IOU);
    float*          out = (float*)d_out;

    const int T = 256;
    // SPP
    spp_kernel<<<(HW * 512 + T - 1) / T, T, 0, stream>>>(feat, X0);
    // weight repacks (BN folded) + pad-row init
    repack_kernel<<<(512 * 2048 + T - 1) / T, T, 0, stream>>>(nW, nG, nB, nM, nV, WN, BN, 512, 2048, 1);
    repack_kernel<<<(256 * 512 + T - 1) / T, T, 0, stream>>>(c1W, c1G, c1B, c1M, c1V, W1, B1, 256, 512, 1);
    repack_kernel<<<(512 * 256 * 9 + T - 1) / T, T, 0, stream>>>(c2W, c2G, c2B, c2M, c2V, W2, B2, 512, 256, 9);
    repack_kernel<<<(256 * 512 + T - 1) / T, T, 0, stream>>>(c3W, c3G, c3B, c3M, c3V, W3, B3, 256, 512, 1);
    repack_kernel<<<(512 * 256 * 9 + T - 1) / T, T, 0, stream>>>(c4W, c4G, c4B, c4M, c4V, W4, B4, 512, 256, 9);
    pred_repack<<<(32 * 512 + T - 1) / T, T, 0, stream>>>(pW, pB, WP, BP);
    zero_pad_rows<<<1, 256, 0, stream>>>(X2, X4);

    // GEMMs: blockDim 128 = 4 waves, each wave a 16x64 tile; waves = (M/16)*25
    auto blocks = [](int M) { return ((M >> 4) * 25 + 3) / 4; };
    gemm1x1<0><<<blocks(512), 128, 0, stream>>>(WN, X0, BN, X1, nullptr, 512, 2048); // neck
    gemm1x1<0><<<blocks(256), 128, 0, stream>>>(W1, X1, B1, X2, nullptr, 256, 512);  // c1
    gemm3x3<<<blocks(512), 128, 0, stream>>>(W2, X2, B2, X3);                        // c2 3x3
    gemm1x1<0><<<blocks(256), 128, 0, stream>>>(W3, X3, B3, X4, nullptr, 256, 512);  // c3
    gemm3x3<<<blocks(512), 128, 0, stream>>>(W4, X4, B4, X5);                        // c4 3x3
    gemm1x1<1><<<blocks(32), 128, 0, stream>>>(WP, X5, BP, nullptr, PR, 32, 512);    // pred

    // decode / score
    decode_kernel<<<(HW + T - 1) / T, T, 0, stream>>>(PR, out, BX, SC, LB);
    // IoU matrix (written after all GEMMs; aliases dead X0/WN/X1 region)
    iou_kernel<<<(HW * HW + T - 1) / T, T, 0, stream>>>(BX, IOU);
    // sort + NMS
    rank_kernel<<<(HW + T - 1) / T, T, 0, stream>>>(SC, OR_);
    zero_keep<<<(HW + T - 1) / T, T, 0, stream>>>(KP);
    nms_kernel<<<NCLS, T, 0, stream>>>(OR_, LB, SC, IOU, KP);
    finalize_keep<<<(HW + T - 1) / T, T, 0, stream>>>(KP, out);
}
